// ContextualClipV1_10041633538759
// MI455X (gfx1250) — matrix operand order repaired and checked by reading, not compiled, not run on hardware
//
#include <hip/hip_runtime.h>
#include <hip/hip_bf16.h>

// Problem constants (match reference)
#define B_     256
#define N_     257
#define D_     768
#define K_     8
#define V_     10000
#define ITERS_ 4

typedef __attribute__((ext_vector_type(16))) __bf16 v16bf;
typedef __attribute__((ext_vector_type(8)))  float  v8f;

static __device__ __forceinline__ float waveReduceSum(float v) {
#pragma unroll
  for (int off = 16; off > 0; off >>= 1)
    v += __shfl_down(v, off, 32);
  return v;
}

// ---------------------------------------------------------------------------
// Kernel 1: gather visual_concepts[topk] and L2-normalize -> initial centroids
// grid = B*K*2 blocks, 256 threads; each block handles one 768-float row.
// cent0 layout: [b][k][c][d]  (row index = b*16 + k*2 + c)
// ---------------------------------------------------------------------------
__global__ __launch_bounds__(256)
void init_centroids_kernel(const float* __restrict__ vc,
                           const int*   __restrict__ topk,
                           float*       __restrict__ cent0) {
  const int row = blockIdx.x;            // b*16 + k*2 + c
  const int b   = row >> 4;
  const int rem = row & 15;
  const int k   = rem >> 1;
  const int c   = rem & 1;
  const int v   = topk[b * K_ + k];
  const float* src = vc + ((size_t)v * 2 + c) * D_;

  const int tid = threadIdx.x;
  float x0 = src[tid];
  float x1 = src[tid + 256];
  float x2 = src[tid + 512];
  float ss = x0 * x0 + x1 * x1 + x2 * x2;
  ss = waveReduceSum(ss);

  __shared__ float warps[8];
  if ((tid & 31) == 0) warps[tid >> 5] = ss;
  __syncthreads();
  float tot = 0.f;
#pragma unroll
  for (int i = 0; i < 8; ++i) tot += warps[i];
  const float inv = 1.f / fmaxf(sqrtf(tot), 1e-12f);

  float* dst = cent0 + (size_t)row * D_;
  dst[tid]       = x0 * inv;
  dst[tid + 256] = x1 * inv;
  dst[tid + 512] = x2 * inv;
}

// ---------------------------------------------------------------------------
// Kernel 2: fused 4-iteration 2-way k-means per batch.  One block per batch.
// 256 threads = 8 waves (wave32).  All matmuls via v_wmma_f32_16x16x32_bf16.
// Outputs: assignments (B,256,K) fp32, cadd[b*K+k] = final centroids[b,k,0,0].
// ---------------------------------------------------------------------------
__global__ __launch_bounds__(256)
void kmeans_kernel(const float* __restrict__ tokens,
                   const float* __restrict__ cent0,
                   float*       __restrict__ assign_out,
                   float*       __restrict__ cadd) {
  __shared__ __bf16 cent [16 * D_];   // 24 KB  [kc][d] current centroids (bf16)
  __shared__ __bf16 cnew [16 * D_];   // 24 KB  [kc][d] un-normalized new centroids
  __shared__ __bf16 wmatT[16 * 256];  //  8 KB  [kc][n] assignment weights (0/1)
  __shared__ float  normsq[16];

  const int b    = blockIdx.x;
  const int tid  = threadIdx.x;
  const int lane = tid & 31;
  const int wave = tid >> 5;
  const int half = lane >> 4;   // 0 or 1
  const int l15  = lane & 15;

  // tp = tokens[b, 1:, :]  (256 x 768)
  const float* tok_b = tokens + (size_t)b * N_ * D_ + D_;

  // Stage initial centroids (fp32 -> bf16) into LDS.
  {
    const float* c0 = cent0 + (size_t)b * 16 * D_;
#pragma unroll
    for (int i = 0; i < 48; ++i) {
      const int idx = tid + i * 256;
      cent[idx] = (__bf16)c0[idx];
    }
  }
  __syncthreads();

  for (int iter = 0; iter < ITERS_; ++iter) {
    if (tid < 16) normsq[tid] = 0.f;

    // ---------------- sim GEMM: sim[n, kc] = sum_d tp[n,d]*cent[kc,d] -------
    // Each wave owns token tiles t = 2*wave, 2*wave+1 (16 tokens each).
#pragma unroll
    for (int tt = 0; tt < 2; ++tt) {
      const int t = wave * 2 + tt;
      const float* rowp = tok_b + (size_t)(t * 16 + l15) * D_;  // A row (M=l15)
      if (tt == 0)  // pull the next tile's row into cache while we compute
        __builtin_prefetch(tok_b + (size_t)((t + 1) * 16 + l15) * D_, 0, 1);
      v8f acc = {};
      for (int s = 0; s < 24; ++s) {          // K = 768 / 32
        const int dbase = s * 32;
        // A tile 16x32 bf16: lane(m=l15,half): e<8  -> d=dbase+8*half+e
        //                                      e>=8 -> d=dbase+16+8*half+(e-8)
        const float* p0 = rowp + dbase + 8 * half;
        const float4 f0 = *(const float4*)(p0);
        const float4 f1 = *(const float4*)(p0 + 4);
        const float4 f2 = *(const float4*)(p0 + 16);
        const float4 f3 = *(const float4*)(p0 + 20);
        v16bf a;
        a[0]=(__bf16)f0.x; a[1]=(__bf16)f0.y; a[2]=(__bf16)f0.z; a[3]=(__bf16)f0.w;
        a[4]=(__bf16)f1.x; a[5]=(__bf16)f1.y; a[6]=(__bf16)f1.z; a[7]=(__bf16)f1.w;
        a[8]=(__bf16)f2.x; a[9]=(__bf16)f2.y; a[10]=(__bf16)f2.z; a[11]=(__bf16)f2.w;
        a[12]=(__bf16)f3.x; a[13]=(__bf16)f3.y; a[14]=(__bf16)f3.z; a[15]=(__bf16)f3.w;
        // B tile 32x16 bf16: lane(N=l15): e<8 -> K(d)=dbase+8*half+e, e>=8 -> dbase+16+8*half+(e-8)
        const __bf16* crow = &cent[l15 * D_ + dbase + 8 * half];
        v16bf bm;
#pragma unroll
        for (int e = 0; e < 16; ++e) bm[e] = crow[e < 8 ? e : e + 8];
        acc = __builtin_amdgcn_wmma_f32_16x16x32_bf16(
            false, a, false, bm, (short)0, acc, false, false);
      }
      // acc[r] = sim[m = 16t + r + 8*half][kc = l15]
      const bool odd = (lane & 1) != 0;
#pragma unroll
      for (int r = 0; r < 8; ++r) {
        const float s0 = acc[r];
        const float s1 = __shfl_xor(s0, 1, 32);   // partner c value
        const int   n  = t * 16 + r + 8 * half;
        // assign = (sim[c=1] > sim[c=0]); odd lanes hold c=1 in s0.
        // weight: odd lane (kc=2k+1) stores assign; even lane stores 1-assign.
        const float w = odd ? ((s0 > s1) ? 1.f : 0.f)
                            : ((s1 > s0) ? 0.f : 1.f);
        wmatT[l15 * 256 + n] = (__bf16)w;         // [kc][n] (transposed)
        if (odd && iter == ITERS_ - 1)            // for odd lanes w == assign
          assign_out[((size_t)b * 256 + n) * K_ + (l15 >> 1)] = w;
      }
    }
    __syncthreads();

    // ------------- centroid GEMM: c[kc,d] = sum_n w[n,kc]*tp[n,d] -----------
    // 48 D-tiles of 16; wave w handles tiles w, w+8, ..., w+40.
#pragma unroll
    for (int jt = 0; jt < 6; ++jt) {
      const int dbase = (wave + jt * 8) * 16;
      v8f acc = {};
      for (int s = 0; s < 8; ++s) {             // K = 256 / 32 over n
        const int nbase = s * 32;
        // A = w^T 16(kc) x 32(n): lane(m=l15,half):
        //   e<8  -> n = nbase + 8*half + e        (contiguous)
        //   e>=8 -> n = nbase + 16 + 8*half + e-8 (contiguous)
        const __bf16* wrow = &wmatT[l15 * 256 + nbase + 8 * half];
        v16bf a, bm;
#pragma unroll
        for (int e = 0; e < 8; ++e)  a[e] = wrow[e];
#pragma unroll
        for (int e = 8; e < 16; ++e) a[e] = wrow[e + 8];
#pragma unroll
        for (int e = 0; e < 16; ++e) {          // B = tp 32(n) x 16(d)
          const int n = nbase + (e < 8 ? e : e + 8) + 8 * half;
          bm[e] = (__bf16)tok_b[(size_t)n * D_ + dbase + l15];
        }
        acc = __builtin_amdgcn_wmma_f32_16x16x32_bf16(
            false, a, false, bm, (short)0, acc, false, false);
      }
#pragma unroll
      for (int r = 0; r < 8; ++r) {
        const int  kc = r + 8 * half;
        const float v = acc[r];
        cnew[kc * D_ + dbase + l15] = (__bf16)v;
        atomicAdd(&normsq[kc], v * v);          // ds_add_f32
      }
    }
    __syncthreads();

    // ------------- normalize rows: cent = cnew / max(||cnew||, eps) --------
#pragma unroll
    for (int i = 0; i < 48; ++i) {
      const int idx = tid + i * 256;
      const int kc  = idx / D_;
      const float inv = 1.f / fmaxf(sqrtf(normsq[kc]), 1e-12f);
      cent[idx] = (__bf16)((float)cnew[idx] * inv);
    }
    if (iter == ITERS_ - 1 && tid < K_) {
      // Only centroids[b,k,c=0,d=0] feeds the output scatter.
      const float inv = 1.f / fmaxf(sqrtf(normsq[tid * 2]), 1e-12f);
      cadd[b * K_ + tid] = (float)cnew[(tid * 2) * D_] * inv;
    }
    __syncthreads();
  }
}

// ---------------------------------------------------------------------------
// Kernel 3a/3b: scatter-add the 2048 scalars into a per-concept delta array.
// ---------------------------------------------------------------------------
__global__ void zero_kernel(float* __restrict__ p, int n) {
  const int i = blockIdx.x * blockDim.x + threadIdx.x;
  if (i < n) p[i] = 0.f;
}

__global__ void scatter_kernel(const int*   __restrict__ topk,
                               const float* __restrict__ cadd,
                               float*       __restrict__ addv) {
  const int i = blockIdx.x * blockDim.x + threadIdx.x;
  if (i < B_ * K_) atomicAdd(&addv[topk[i]], cadd[i]);
}

// ---------------------------------------------------------------------------
// Kernel 4: vc_new = l2norm(visual_concepts with [v,0,0] += addv[v])
// grid = V*2 blocks (one per (v,c) row), 256 threads.
// ---------------------------------------------------------------------------
__global__ __launch_bounds__(256)
void norm_vc_kernel(const float* __restrict__ vc,
                    const float* __restrict__ addv,
                    float*       __restrict__ out) {
  const int row = blockIdx.x;          // v*2 + c
  const int c   = row & 1;
  const int v   = row >> 1;
  const float* src = vc + (size_t)row * D_;
  const int tid = threadIdx.x;

  float x0 = src[tid];
  float x1 = src[tid + 256];
  float x2 = src[tid + 512];
  if (c == 0 && tid == 0) x0 += addv[v];

  float ss = x0 * x0 + x1 * x1 + x2 * x2;
  ss = waveReduceSum(ss);
  __shared__ float warps[8];
  if ((tid & 31) == 0) warps[tid >> 5] = ss;
  __syncthreads();
  float tot = 0.f;
#pragma unroll
  for (int i = 0; i < 8; ++i) tot += warps[i];
  const float inv = 1.f / fmaxf(sqrtf(tot), 1e-12f);

  float* dst = out + (size_t)row * D_;
  dst[tid]       = x0 * inv;
  dst[tid + 256] = x1 * inv;
  dst[tid + 512] = x2 * inv;
}

// ---------------------------------------------------------------------------
extern "C" void kernel_launch(void* const* d_in, const int* in_sizes, int n_in,
                              void* d_out, int out_size, void* d_ws, size_t ws_size,
                              hipStream_t stream) {
  const float* tokens = (const float*)d_in[0];   // (B, N, D) fp32
  const float* vc     = (const float*)d_in[1];   // (V, 2, D) fp32
  const int*   topk   = (const int*)d_in[2];     // (B, K) int32

  float* out         = (float*)d_out;
  float* assignments = out;                                // B*256*K
  float* vc_new      = out + (size_t)B_ * 256 * K_;        // V*2*D

  // Workspace layout (fp32): cent0 | cadd | addv
  float* cent0 = (float*)d_ws;                             // B*16*D = 3,145,728
  float* cadd  = cent0 + (size_t)B_ * 16 * D_;             // B*K    = 2,048
  float* addv  = cadd + (size_t)B_ * K_;                   // V      = 10,000

  init_centroids_kernel<<<B_ * K_ * 2, 256, 0, stream>>>(vc, topk, cent0);
  kmeans_kernel<<<B_, 256, 0, stream>>>(tokens, cent0, assignments, cadd);
  zero_kernel<<<(V_ + 255) / 256, 256, 0, stream>>>(addv, V_);
  scatter_kernel<<<(B_ * K_ + 255) / 256, 256, 0, stream>>>(topk, cadd, addv);
  norm_vc_kernel<<<V_ * 2, 256, 0, stream>>>(vc, addv, vc_new);
}
